// multi_dRNN_with_dilations_20383914787300
// MI455X (gfx1250) — matrix operand order repaired and checked by reading, not compiled, not run on hardware
//
#include <hip/hip_runtime.h>
#include <hip/hip_bf16.h>
#include <stdint.h>

// ---------- vector types ----------
typedef __bf16  v16bf __attribute__((ext_vector_type(16)));
typedef float   v8f   __attribute__((ext_vector_type(8)));
typedef int     i32x4 __attribute__((ext_vector_type(4)));
typedef int     i32x8 __attribute__((ext_vector_type(8)));
typedef unsigned int u32x4 __attribute__((ext_vector_type(4)));

union TileU { i32x4 q[2]; v16bf v; };

// ---------- activations ----------
__device__ __forceinline__ float sigm(float x) {
    return 1.0f / (1.0f + __expf(-x));
}
__device__ __forceinline__ float tanh_apx(float x) {
    float t = __expf(-2.0f * fabsf(x));
    float r = (1.0f - t) / (1.0f + t);
    return copysignf(r, x);
}

// ---------- TDM: load one [4 blocks x 32 rows x 32 cols] bf16 W slice to LDS ----------
// Tensor: W slice starting at gsrc, row stride K elements, gate-block stride 256*K.
// LDS pad: 4 DWORDs every 16 DWORDs (each 64B row -> 80B stride, conflict-free reads).
template<int KK>
__device__ __forceinline__ void tdm_load_w(unsigned lds_off, const __bf16* gsrc) {
    unsigned long long ga = (unsigned long long)(uintptr_t)gsrc;
    u32x4 g0;
    g0[0] = 1u;                                   // count=1, user mode
    g0[1] = lds_off;                              // LDS byte address
    g0[2] = (unsigned)ga;                         // global addr [31:0]
    g0[3] = (unsigned)((ga >> 32) & 0x1FFFFFFull) // global addr [56:32]
          | (2u << 30);                           // type=2 ("image")
    i32x8 g1;
    g1[0] = (int)((1u << 16)                      // data_size = 2B
                | (1u << 20)                      // pad_enable
                | (3u << 22)                      // pad_interval: 16 DWORDs
                | (3u << 25));                    // pad_amount: 4 DWORDs
    g1[1] = (int)(((unsigned)KK & 0xFFFFu) << 16);// tensor_dim0[15:0]
    g1[2] = (int)(256u << 16);                    // tensor_dim0[31:16]=0 | tensor_dim1[15:0]=256
    g1[3] = (int)(32u << 16);                     // tensor_dim1[31:16]=0 | tile_dim0=32
    g1[4] = (int)(32u | (4u << 16));              // tile_dim1=32 | tile_dim2=4
    g1[5] = KK;                                   // tensor_dim0_stride[31:0] = K
    unsigned d1s = 256u * (unsigned)KK;           // tensor_dim1_stride = block hop
    g1[6] = (int)((d1s & 0xFFFFu) << 16);         // dim0_stride[47:32]=0 | dim1_stride[15:0]
    g1[7] = (int)(d1s >> 16);                     // dim1_stride[47:16]
    i32x4 g2; g2[0] = 4; g2[1] = 0; g2[2] = 0; g2[3] = 0;   // tensor_dim2=4
    i32x4 g3; g3[0] = 0; g3[1] = 0; g3[2] = 0; g3[3] = 0;
    i32x8 g4; g4[0]=0; g4[1]=0; g4[2]=0; g4[3]=0; g4[4]=0; g4[5]=0; g4[6]=0; g4[7]=0;
    __builtin_amdgcn_tensor_load_to_lds(g0, g1, g2, g3, g4, 0);
}

// ---------- fused dilated-LSTM layer scan ----------
// Grid: C/16 workgroups of 256 threads (8 waves). WG owns 16 chains; wave wv owns
// h columns [wv*32, wv*32+32). gates = [x_t | h] @ [W_ih|W_hh]^T + b via bf16 WMMA.
template<int DIN, int RATE, bool LAST>
__global__ __launch_bounds__(256) void lstm_scan(
    const __bf16* __restrict__ xin,   // [131072, DIN] bf16
    const __bf16* __restrict__ wcat,  // [1024, K] bf16 (W_ih | W_hh)
    const float*  __restrict__ bias,  // [1024] f32 (b_ih + b_hh)
    __bf16*       __restrict__ hbf,   // [131072, 256] bf16 out (next layer input)
    float*        __restrict__ hf32)  // [131072, 256] f32 out (d_out) if LAST
{
    constexpr int K      = DIN + 256;
    constexpr int C      = RATE * 128;
    constexpr int NSTEP  = 1024 / RATE;
    constexpr int NC     = K / 32;        // K-chunks of 32
    constexpr int APITCH = K + 8;         // A row pitch (elements), kills bank conflicts
    constexpr int A_BYTES = 16 * APITCH * 2;
    constexpr int WROWB  = 80;            // padded LDS bytes per 32-col row
    constexpr int WBUF   = 128 * WROWB;   // one chunk buffer per wave (10240 B)

    __shared__ __align__(16) unsigned char smem[A_BYTES + 8 * 2 * WBUF];

    const int tid  = threadIdx.x;
    const int lane = tid & 31;
    const int wv   = tid >> 5;
    const int wg   = blockIdx.x;
    const int ln15 = lane & 15;
    const int hi   = (lane >> 4) & 1;

    unsigned char* sA = smem;
    const unsigned wldsbase = (unsigned)(A_BYTES + wv * 2 * WBUF);

    // zero the h region of A_lds (cols DIN..DIN+255)
    for (int i = tid; i < 16 * 128; i += 256) {
        int r = i >> 7, c = (i & 127) * 2;
        *(unsigned*)(sA + ((size_t)r * APITCH + DIN + c) * 2) = 0u;
    }

    // per-lane bias for the 8 gate sub-tiles (nt = gate*2 + half)
    float bias8[8];
#pragma unroll
    for (int nt = 0; nt < 8; ++nt)
        bias8[nt] = bias[(nt >> 1) * 256 + wv * 32 + (nt & 1) * 16 + ln15];

    v8f cst[2];                              // c-state in WMMA C/D layout
#pragma unroll
    for (int h2 = 0; h2 < 2; ++h2)
#pragma unroll
        for (int d = 0; d < 8; ++d) cst[h2][d] = 0.0f;

    const __bf16* gw = wcat + (size_t)wv * 32 * K;   // wave's first gate row

    for (int s = 0; s < NSTEP; ++s) {
        // ---- stage x_t into A_lds cols [0, DIN) : rows are contiguous in memory
        {
            constexpr int CH = DIN / 8;            // 16B pieces per row
            const __bf16* xrow = xin + ((size_t)s * C + wg * 16) * DIN;
            for (int idx = tid; idx < 16 * CH; idx += 256) {
                int r = idx / CH, cc = (idx % CH) * 8;
                *(i32x4*)(sA + ((size_t)r * APITCH + cc) * 2) =
                    *(const i32x4*)(xrow + (size_t)r * DIN + cc);
            }
        }

        // ---- prime TDM double buffer (per-wave private W stream from L2)
        asm volatile("s_wait_dscnt 0x0" ::: "memory");   // WAR vs prev step's LDS reads
        tdm_load_w<K>(wldsbase + 0,    gw + 0);
        tdm_load_w<K>(wldsbase + WBUF, gw + 32);

        __syncthreads();   // x + prev-step h visible to all waves

        v8f acc[8];
#pragma unroll
        for (int nt = 0; nt < 8; ++nt) {
            v8f a;
#pragma unroll
            for (int d = 0; d < 8; ++d) a[d] = bias8[nt];
            acc[nt] = a;
        }

#pragma unroll
        for (int kc = 0; kc < NC; ++kc) {
            // TDM completes in order per wave: oldest outstanding is chunk kc.
            if (kc == NC - 1) __builtin_amdgcn_s_wait_tensorcnt(0);
            else              __builtin_amdgcn_s_wait_tensorcnt(1);

            // A tile: M=16, K=32 slab kc (16-bit A layout: lanes 16-31 at +16B, v4..7 at +32B)
            TileU A;
            {
                unsigned char* pa = sA + ((size_t)ln15 * APITCH + kc * 32) * 2 + (hi ? 16 : 0);
                A.q[0] = *(const i32x4*)pa;
                A.q[1] = *(const i32x4*)(pa + 32);
            }
            unsigned char* wb = smem + wldsbase + (kc & 1) * WBUF;
#pragma unroll
            for (int nt = 0; nt < 8; ++nt) {
                int rowin = (nt >> 1) * 32 + (nt & 1) * 16 + ln15;
                unsigned char* pb = wb + rowin * WROWB + (hi ? 16 : 0);
                TileU B;
                B.q[0] = *(const i32x4*)pb;
                B.q[1] = *(const i32x4*)(pb + 32);
                acc[nt] = __builtin_amdgcn_wmma_f32_16x16x32_bf16(
                    false, A.v, false, B.v, (short)0, acc[nt], false, false);
            }
            if (kc + 2 < NC) {
                asm volatile("s_wait_dscnt 0x0" ::: "memory");  // reads of this buffer done
                tdm_load_w<K>(wldsbase + (kc & 1) * WBUF, gw + (kc + 2) * 32);
            }
        }

        __syncthreads();   // all waves finished reading A_lds

        // ---- LSTM cell (torch gate order i,f,g,o), write h to LDS + global
        size_t orow0 = (size_t)s * C + wg * 16;
#pragma unroll
        for (int hf = 0; hf < 2; ++hf) {
            v8f iv = acc[0 + hf], fv = acc[2 + hf], gv = acc[4 + hf], ov = acc[6 + hf];
            int j = wv * 32 + hf * 16 + ln15;
#pragma unroll
            for (int d = 0; d < 8; ++d) {
                float cn = sigm(fv[d]) * cst[hf][d] + sigm(iv[d]) * tanh_apx(gv[d]);
                float hh = sigm(ov[d]) * tanh_apx(cn);
                cst[hf][d] = cn;
                int m = d + hi * 8;
                *(__bf16*)(sA + ((size_t)m * APITCH + DIN + j) * 2) = (__bf16)hh;
                size_t go = (orow0 + m) * 256 + j;
                if (LAST) hf32[go] = hh;
                else      hbf[go]  = (__bf16)hh;
            }
        }
        // next iteration's barrier covers h visibility; x/W regions are disjoint or per-wave
    }
}

// ---------- prep kernels ----------
__global__ void k_cvt_bf16(const float* __restrict__ in, __bf16* __restrict__ out, int n) {
    int i = blockIdx.x * blockDim.x + threadIdx.x;
    int stride = gridDim.x * blockDim.x;
    for (; i < n; i += stride) out[i] = (__bf16)in[i];
}

__global__ void k_wcat(const float* __restrict__ wih, const float* __restrict__ whh,
                       __bf16* __restrict__ wcat, int din) {
    int row = blockIdx.x;
    int K = din + 256;
    for (int c = threadIdx.x; c < K; c += blockDim.x) {
        float v = (c < din) ? wih[(size_t)row * din + c] : whh[(size_t)row * 256 + (c - din)];
        wcat[(size_t)row * K + c] = (__bf16)v;
    }
}

__global__ void k_bias(const float* __restrict__ a, const float* __restrict__ b,
                       float* __restrict__ o) {
    int i = blockIdx.x * blockDim.x + threadIdx.x;
    if (i < 1024) o[i] = a[i] + b[i];
}

// ---------- launch ----------
extern "C" void kernel_launch(void* const* d_in, const int* in_sizes, int n_in,
                              void* d_out, int out_size, void* d_ws, size_t ws_size,
                              hipStream_t stream) {
    (void)in_sizes; (void)n_in; (void)out_size; (void)ws_size;
    const float* x = (const float*)d_in[0];
    const float* Wih[3] = {(const float*)d_in[1], (const float*)d_in[5], (const float*)d_in[9]};
    const float* Whh[3] = {(const float*)d_in[2], (const float*)d_in[6], (const float*)d_in[10]};
    const float* bih[3] = {(const float*)d_in[3], (const float*)d_in[7], (const float*)d_in[11]};
    const float* bhh[3] = {(const float*)d_in[4], (const float*)d_in[8], (const float*)d_in[12]};

    char* ws = (char*)d_ws;
    size_t off = 0;
    auto take = [&](size_t bytes) {
        char* p = ws + off;
        off = (off + bytes + 255) & ~(size_t)255;
        return p;
    };
    __bf16* x0 = (__bf16*)take(131072ull * 128 * 2);
    __bf16* h1 = (__bf16*)take(131072ull * 256 * 2);
    __bf16* h2 = (__bf16*)take(131072ull * 256 * 2);
    __bf16* w0 = (__bf16*)take(1024ull * 384 * 2);
    __bf16* w1 = (__bf16*)take(1024ull * 512 * 2);
    __bf16* w2 = (__bf16*)take(1024ull * 512 * 2);
    float*  b0 = (float*)take(1024 * 4);
    float*  b1 = (float*)take(1024 * 4);
    float*  b2 = (float*)take(1024 * 4);

    k_cvt_bf16<<<4096, 256, 0, stream>>>(x, x0, 131072 * 128);
    k_wcat<<<1024, 256, 0, stream>>>(Wih[0], Whh[0], w0, 128);
    k_wcat<<<1024, 256, 0, stream>>>(Wih[1], Whh[1], w1, 256);
    k_wcat<<<1024, 256, 0, stream>>>(Wih[2], Whh[2], w2, 256);
    k_bias<<<4, 256, 0, stream>>>(bih[0], bhh[0], b0);
    k_bias<<<4, 256, 0, stream>>>(bih[1], bhh[1], b1);
    k_bias<<<4, 256, 0, stream>>>(bih[2], bhh[2], b2);

    float* out = (float*)d_out;
    lstm_scan<128, 1, false><<<8,  256, 0, stream>>>(x0, w0, b0, h1, out);
    lstm_scan<256, 2, false><<<16, 256, 0, stream>>>(h1, w1, b1, h2, out);
    lstm_scan<256, 4, true ><<<32, 256, 0, stream>>>(h2, w2, b2, h1, out);
}
